// MIND_45466523796149
// MI455X (gfx1250) — matrix operand order repaired and checked by reading, not compiled, not run on hardware
//
#include <hip/hip_runtime.h>
#include <hip/hip_bf16.h>
#include <stdint.h>

typedef __attribute__((ext_vector_type(16))) __bf16 bf16x16;
typedef __attribute__((ext_vector_type(8)))  float  f32x8;
typedef unsigned short ushort_t;

#define B_ 1024
#define S_ 50
#define H_ 128
#define K_ 4
#define V_ 100000
#define KS_ (K_*S_)           // 200
#define NCHUNK 50
#define NTPC 125              // n-tiles (of 16 cols) per chunk: 100000/16/50

union AFrag { bf16x16 v; uint4 q[2]; };

__device__ __forceinline__ ushort_t f2bf(float f) {
  unsigned int u = __float_as_uint(f);
  u += 0x7fffu + ((u >> 16) & 1u);      // round-to-nearest-even
  return (ushort_t)(u >> 16);
}
__device__ __forceinline__ float bf2f(ushort_t u) {
  return __uint_as_float(((unsigned int)u) << 16);
}

// ---------- generic convert / copy ----------
__global__ void k_f32_to_bf16(const float* __restrict__ src, ushort_t* __restrict__ dst, int n) {
  int i = blockIdx.x * blockDim.x + threadIdx.x;
  if (i < n) dst[i] = f2bf(src[i]);
}
__global__ void k_copy_f32(const float* __restrict__ src, float* __restrict__ dst, int n) {
  int i = blockIdx.x * blockDim.x + threadIdx.x;
  if (i < n) dst[i] = src[i];
}

// ---------- item_e gather (emb row 0 zeroed) ----------
__global__ void k_item_gather(const int* __restrict__ item, const float* __restrict__ emb,
                              float* __restrict__ item_e) {
  int i = blockIdx.x * blockDim.x + threadIdx.x;   // B_*H_
  int b = i >> 7, hh = i & 127;
  int idx = item[b];
  item_e[i] = (idx == 0) ? 0.0f : emb[(size_t)idx * H_ + hh];
}

// ---------- h = gather(seq) @ W^T via bf16 WMMA ----------
// one wave per 16-row tile of (b,s); K=128 in 4 chunks; N=128 in 8 tiles
__global__ void k_h_wmma(const int* __restrict__ item_seq, const ushort_t* __restrict__ emb_bf,
                         const ushort_t* __restrict__ W_bf, float* __restrict__ h) {
  int lane = threadIdx.x;
  int row = lane & 15, half = lane >> 4;
  int m = blockIdx.x * 16 + row;
  int idx = item_seq[m];
  AFrag a[4];
  if (idx == 0) {
    uint4 z = make_uint4(0u, 0u, 0u, 0u);
#pragma unroll
    for (int kc = 0; kc < 4; ++kc) { a[kc].q[0] = z; a[kc].q[1] = z; }
  } else {
    const ushort_t* rp = emb_bf + (size_t)idx * H_;
#pragma unroll
    for (int kc = 0; kc < 4; ++kc) {
      const ushort_t* base = rp + kc * 32;
      a[kc].q[0] = *(const uint4*)(base + half * 8);        // K = half*8 .. +7
      a[kc].q[1] = *(const uint4*)(base + 16 + half * 8);   // K = 16+half*8 .. +7
    }
  }
  int mrow = blockIdx.x * 16 + half * 8;
#pragma unroll
  for (int nt = 0; nt < 8; ++nt) {
    f32x8 c = {0.f, 0.f, 0.f, 0.f, 0.f, 0.f, 0.f, 0.f};
#pragma unroll
    for (int kc = 0; kc < 4; ++kc) {
      const ushort_t* bp = W_bf + (size_t)(nt * 16 + row) * H_ + kc * 32 + half * 8;
      AFrag bfr;
      bfr.q[0] = *(const uint4*)bp;
      bfr.q[1] = *(const uint4*)(bp + 16);
      c = __builtin_amdgcn_wmma_f32_16x16x32_bf16(false, a[kc].v, false, bfr.v,
                                                  (short)0, c, false, false);
    }
#pragma unroll
    for (int r = 0; r < 8; ++r)
      h[(size_t)(mrow + r) * H_ + nt * 16 + row] = c[r];
  }
}

// ---------- softmax over batch axis (axis=0), then mask ----------
__global__ void k_softmax_b(const float* __restrict__ cw, const float* __restrict__ mask,
                            float* __restrict__ sw) {
  __shared__ float red[256];
  int col = blockIdx.x;            // k*S + s
  int s = col % S_;
  int tid = threadIdx.x;
  float v[4]; float mx = -3.4e38f;
#pragma unroll
  for (int i = 0; i < 4; ++i) {
    v[i] = cw[(size_t)(tid + i * 256) * KS_ + col];
    mx = fmaxf(mx, v[i]);
  }
  red[tid] = mx; __syncthreads();
  for (int st = 128; st; st >>= 1) { if (tid < st) red[tid] = fmaxf(red[tid], red[tid + st]); __syncthreads(); }
  mx = red[0]; __syncthreads();
  float sum = 0.f;
#pragma unroll
  for (int i = 0; i < 4; ++i) { v[i] = __expf(v[i] - mx); sum += v[i]; }
  red[tid] = sum; __syncthreads();
  for (int st = 128; st; st >>= 1) { if (tid < st) red[tid] += red[tid + st]; __syncthreads(); }
  float inv = 1.0f / red[0];
#pragma unroll
  for (int i = 0; i < 4; ++i) {
    int b = tid + i * 256;
    float msk = mask[(size_t)b * S_ + s];
    sw[(size_t)b * KS_ + col] = (msk == 0.f) ? 0.f : v[i] * inv;
  }
}

// ---------- capsule step: c_j, squash, delta (or final user_emb) ----------
__global__ void k_caps(const float* __restrict__ h, const float* __restrict__ sw,
                       float* __restrict__ cw, float* __restrict__ out_user, int final_iter) {
  __shared__ float cs[32];
  int bk = blockIdx.x; int b = bk >> 2; int k = bk & 3;
  int lane = threadIdx.x;          // 0..31 -> j
  const float* hb = h + (size_t)b * S_ * H_ + k * 32;
  const float* swp = sw + (size_t)b * KS_ + k * S_;
  float c = 0.f;
  for (int s = 0; s < S_; ++s) c += swp[s] * hb[(size_t)s * H_ + lane];
  float t = c * c;
#pragma unroll
  for (int off = 16; off; off >>= 1) t += __shfl_xor(t, off, 32);
  float norm = 4.f * t;            // cap is 4-replicated: ||cap||^2 = 4*sum c_j^2
  float scalar = norm / (1.f + norm) * rsqrtf(norm + 1e-9f);
  if (final_iter) {
    float val = scalar * c;
    float4 o = make_float4(val, val, val, val);
    *(float4*)(out_user + (size_t)bk * H_ + lane * 4) = o;
  } else {
    cs[lane] = c; __syncthreads();
    float* cwp = cw + (size_t)b * KS_ + k * S_;
    for (int s = lane; s < S_; s += 32) {
      float d = 0.f;
#pragma unroll
      for (int j = 0; j < 32; ++j) d += hb[(size_t)s * H_ + j] * cs[j];
      cwp[s] += 4.f * scalar * d;
    }
  }
}

// ---------- cos, argmax over K, select best interest (store bf16) ----------
__global__ void k_cos_best(const float* __restrict__ user, const float* __restrict__ item_e,
                           ushort_t* __restrict__ best_bf) {
  __shared__ float red[4][128];
  __shared__ int kb;
  int b = blockIdx.x, t = threadIdx.x;
  float ie = item_e[(size_t)b * H_ + t];
#pragma unroll
  for (int k = 0; k < 4; ++k) red[k][t] = user[((size_t)b * 4 + k) * H_ + t] * ie;
  __syncthreads();
  for (int st = 64; st; st >>= 1) {
    if (t < st) {
#pragma unroll
      for (int k = 0; k < 4; ++k) red[k][t] += red[k][t + st];
    }
    __syncthreads();
  }
  if (t == 0) {
    int best = 0; float bv = red[0][0];
#pragma unroll
    for (int k = 1; k < 4; ++k) if (red[k][0] > bv) { bv = red[k][0]; best = k; }
    kb = best;
  }
  __syncthreads();
  best_bf[(size_t)b * H_ + t] = f2bf(user[((size_t)b * 4 + kb) * H_ + t]);
}

// ---------- fused scores GEMM (bf16 WMMA) + per-row partial sum of exp ----------
// grid (64 m-tiles, 50 n-chunks), 1 wave each; scores are tiny so unshifted exp is safe
__global__ void k_scores(const ushort_t* __restrict__ best_bf, const ushort_t* __restrict__ emb_bf,
                         float* __restrict__ psm) {
  __shared__ float red[16][16];    // [row][col-lane]
  int lane = threadIdx.x, row = lane & 15, half = lane >> 4;
  int mtile = blockIdx.x, chunk = blockIdx.y;
  AFrag a[4];
  const ushort_t* ap = best_bf + (size_t)(mtile * 16 + row) * H_;
#pragma unroll
  for (int kc = 0; kc < 4; ++kc) {
    a[kc].q[0] = *(const uint4*)(ap + kc * 32 + half * 8);
    a[kc].q[1] = *(const uint4*)(ap + kc * 32 + 16 + half * 8);
  }
  float sm[8];
#pragma unroll
  for (int r = 0; r < 8; ++r) sm[r] = 0.f;
  int n0 = chunk * (NTPC * 16);
  for (int t = 0; t < NTPC; ++t) {
    int col = n0 + t * 16 + row;
    const ushort_t* bp = emb_bf + (size_t)col * H_ + half * 8;
    f32x8 c = {0.f, 0.f, 0.f, 0.f, 0.f, 0.f, 0.f, 0.f};
#pragma unroll
    for (int kc = 0; kc < 4; ++kc) {
      AFrag bfr;
      bfr.q[0] = *(const uint4*)(bp + kc * 32);
      bfr.q[1] = *(const uint4*)(bp + kc * 32 + 16);
      c = __builtin_amdgcn_wmma_f32_16x16x32_bf16(false, a[kc].v, false, bfr.v,
                                                  (short)0, c, false, false);
    }
#pragma unroll
    for (int r = 0; r < 8; ++r) sm[r] += __expf(c[r]);
  }
#pragma unroll
  for (int r = 0; r < 8; ++r) red[half * 8 + r][row] = sm[r];
  __syncthreads();
  if (lane < 16) {
    float tot = 0.f;
#pragma unroll
    for (int cidx = 0; cidx < 16; ++cidx) tot += red[lane][cidx];
    psm[(size_t)(mtile * 16 + lane) * NCHUNK + chunk] = tot;
  }
}

// ---------- score at target item (same bf16 precision as GEMM) ----------
__global__ void k_sitem(const int* __restrict__ item, const ushort_t* __restrict__ best_bf,
                        const ushort_t* __restrict__ emb_bf, float* __restrict__ s_item) {
  __shared__ float red[128];
  int b = blockIdx.x, t = threadIdx.x;
  int idx = item[b];
  red[t] = bf2f(best_bf[(size_t)b * H_ + t]) * bf2f(emb_bf[(size_t)idx * H_ + t]);
  __syncthreads();
  for (int st = 64; st; st >>= 1) { if (t < st) red[t] += red[t + st]; __syncthreads(); }
  if (t == 0) s_item[b] = red[0];
}

// ---------- final loss = -mean(s_item - log(sum exp)) ----------
__global__ void k_loss(const float* __restrict__ psm, const float* __restrict__ s_item,
                       float* __restrict__ out) {
  __shared__ float red[256];
  int t = threadIdx.x;
  float acc = 0.f;
  for (int b = t; b < B_; b += 256) {
    float tot = 0.f;
    for (int c = 0; c < NCHUNK; ++c) tot += psm[(size_t)b * NCHUNK + c];
    acc += s_item[b] - logf(tot);
  }
  red[t] = acc; __syncthreads();
  for (int st = 128; st; st >>= 1) { if (t < st) red[t] += red[t + st]; __syncthreads(); }
  if (t == 0) out[(size_t)B_ * K_ * H_] = -red[0] / (float)B_;
}

extern "C" void kernel_launch(void* const* d_in, const int* in_sizes, int n_in,
                              void* d_out, int out_size, void* d_ws, size_t ws_size,
                              hipStream_t stream) {
  const int*   item_seq = (const int*)  d_in[0];
  const float* mask     = (const float*)d_in[1];
  const int*   item     = (const int*)  d_in[2];
  const float* emb      = (const float*)d_in[3];
  const float* W        = (const float*)d_in[4];
  const float* cw0      = (const float*)d_in[5];
  float* out = (float*)d_out;
  char* ws = (char*)d_ws;

  // workspace layout (all 256B aligned)
  size_t off = 0;
  ushort_t* emb_bf  = (ushort_t*)(ws + off); off += (size_t)V_ * H_ * 2;        // 25.6 MB
  ushort_t* W_bf    = (ushort_t*)(ws + off); off += (size_t)H_ * H_ * 2;        // 32 KB
  float*    h       = (float*)   (ws + off); off += (size_t)B_ * S_ * H_ * 4;   // 26.2 MB
  float*    item_e  = (float*)   (ws + off); off += (size_t)B_ * H_ * 4;        // 0.5 MB
  float*    cw      = (float*)   (ws + off); off += (size_t)B_ * KS_ * 4;       // 0.8 MB
  float*    sw      = (float*)   (ws + off); off += (size_t)B_ * KS_ * 4;       // 0.8 MB
  ushort_t* best_bf = (ushort_t*)(ws + off); off += (size_t)B_ * H_ * 2;        // 0.25 MB
  float*    psm     = (float*)   (ws + off); off += (size_t)B_ * NCHUNK * 4;    // 0.2 MB
  float*    s_item  = (float*)   (ws + off); off += (size_t)B_ * 4;

  // precision conversion + setup
  k_f32_to_bf16<<<(V_ * H_ + 255) / 256, 256, 0, stream>>>(emb, emb_bf, V_ * H_);
  k_f32_to_bf16<<<(H_ * H_ + 255) / 256, 256, 0, stream>>>(W, W_bf, H_ * H_);
  k_copy_f32<<<(B_ * KS_ + 255) / 256, 256, 0, stream>>>(cw0, cw, B_ * KS_);
  k_item_gather<<<(B_ * H_ + 255) / 256, 256, 0, stream>>>(item, emb, item_e);

  // h = gather(seq_emb) @ W^T   (WMMA)
  k_h_wmma<<<(B_ * S_) / 16, 32, 0, stream>>>(item_seq, emb_bf, W_bf, h);

  // dynamic routing (softmax over batch axis each iteration)
  for (int it = 0; it < 3; ++it) {
    k_softmax_b<<<KS_, 256, 0, stream>>>(cw, mask, sw);
    k_caps<<<B_ * K_, 32, 0, stream>>>(h, sw, cw, out, it == 2);
  }

  // best interest selection
  k_cos_best<<<B_, 128, 0, stream>>>(out, item_e, best_bf);

  // scores GEMM + streamed softmax denominator (WMMA)
  k_scores<<<dim3(B_ / 16, NCHUNK), 32, 0, stream>>>(best_bf, emb_bf, psm);
  k_sitem<<<B_, 128, 0, stream>>>(item, best_bf, emb_bf, s_item);
  k_loss<<<1, 256, 0, stream>>>(psm, s_item, out);
}
